// KANLayerBSpline_43147241456055
// MI455X (gfx1250) — matrix operand order repaired and checked by reading, not compiled, not run on hardware
//
#include <hip/hip_runtime.h>

typedef __attribute__((ext_vector_type(16))) _Float16 v16h;
typedef __attribute__((ext_vector_type(4)))  _Float16 v4h;
typedef __attribute__((ext_vector_type(8)))  float    v8f;
typedef __attribute__((ext_vector_type(4)))  unsigned int u32x4;
typedef __attribute__((ext_vector_type(8)))  unsigned int u32x8;

#define BDIM 16384
#define DDIM 512
#define ODIM 64
#define KDIM 4096   // D * N_CTRL
#define GROUP_CHUNKS 8          // K-chunks (of 32) staged per TDM transfer
#define GROUP_ELEMS  (GROUP_CHUNKS * 32)   // 256 f16 per coeff row per group
#define NGROUPS      (KDIM / GROUP_ELEMS)  // 16

// Branchless evaluation of the 8 cubic B-spline basis functions for plain
// uniform knots linspace(-1,1,12): one-hot cell indicator + FMA reconstruction.
// Matches reference Cox-de Boor half-open intervals (0 at x==-1 and x>=1).
__device__ __forceinline__ void bspline8(float xv, float out8[8]) {
    float xc = fminf(fmaxf(xv, -1.0f), 1.0f);   // v_med3
    float t  = (xc + 1.0f) * 5.5f;              // 1/h = 5.5, t in [0,11]
    float tf = floorf(t);
    float u  = t - tf;
    float u2 = u * u, u3 = u2 * u;
    float om = 1.0f - u;
    const float sixth = 1.0f / 6.0f;
    float w0 = om * om * om * sixth;                               // m = j-3
    float w1 = (3.0f * u3 - 6.0f * u2 + 4.0f) * sixth;             // m = j-2
    float w2 = (-3.0f * u3 + 3.0f * u2 + 3.0f * u + 1.0f) * sixth; // m = j-1
    float w3 = u3 * sixth;                                         // m = j
    float g[11];
#pragma unroll
    for (int q = 0; q < 11; ++q)
        g[q] = (tf == (float)q) ? 1.0f : 0.0f;
#pragma unroll
    for (int m = 0; m < 8; ++m)
        out8[m] = w3 * g[m] + w2 * g[m + 1] + w1 * g[m + 2] + w0 * g[m + 3];
}

// coeff fp32 [O, D, 8] -> f16 [O, K] (same flat order), 4 elems/thread.
__global__ __launch_bounds__(256) void kan_cvt_coeff_f16(const float4* __restrict__ in,
                                                         v4h* __restrict__ outh) {
    int i = blockIdx.x * blockDim.x + threadIdx.x;   // ODIM*KDIM/4 threads
    float4 f = in[i];
    v4h h;
    h[0] = (_Float16)f.x; h[1] = (_Float16)f.y;
    h[2] = (_Float16)f.z; h[3] = (_Float16)f.w;
    outh[i] = h;
}

// Issue one TDM 2D tile load: coeffh rows o=0..63, columns [g*256, g*256+256)
// (512 B contiguous per row, row stride 4096 elements) -> LDS, rows packed
// 512 B apart. D# per CDNA5 ISA.
__device__ __forceinline__ void tdm_load_group(const _Float16* coeffh, int g, void* ldsDst) {
    unsigned long long ga = (unsigned long long)(uintptr_t)coeffh +
                            (unsigned long long)g * (GROUP_ELEMS * 2u);
    unsigned int lo = (unsigned int)(uintptr_t)ldsDst;  // addrspace(3) byte offset
    u32x4 d0;
    d0[0] = 1u;                                   // count=1 (valid user D#)
    d0[1] = lo;                                   // lds_addr
    d0[2] = (unsigned int)ga;                     // global_addr[31:0]
    d0[3] = (unsigned int)(ga >> 32) | (2u << 30);// global_addr[56:32] | type=2
    u32x8 d1;
    d1[0] = 0x00010000u;                          // wg_mask=0, data_size=2B, no flags
    d1[1] = (KDIM & 0xFFFFu) << 16;               // tensor_dim0[15:0]=4096 (lo half)
    d1[2] = (KDIM >> 16) | (ODIM << 16);          // tensor_dim0 hi | tensor_dim1=64 lo
    d1[3] = (ODIM >> 16) | (GROUP_ELEMS << 16);   // tensor_dim1 hi | tile_dim0=256
    d1[4] = (unsigned int)ODIM;                   // tile_dim1=64, tile_dim2=0
    d1[5] = (unsigned int)KDIM;                   // tensor_dim0_stride lo (elements)
    d1[6] = 0u;                                   // stride0 hi | stride1 lo (unused)
    d1[7] = 0u;
    asm volatile("tensor_load_to_lds %0, %1" :: "s"(d0), "s"(d1) : "memory");
}

// One wave = 16-row M tile x all 4 N tiles (N = 64). Block = 8 waves = 128 rows.
// USE_TDM: coeff f16 staged to LDS by the Tensor Data Mover (double buffered).
template <bool USE_TDM>
__global__ __launch_bounds__(256) void kan_wmma_kernel(const float* __restrict__ x,
                                                       const float* __restrict__ coeff,
                                                       const _Float16* __restrict__ coeffh,
                                                       const float* __restrict__ bias,
                                                       float* __restrict__ out) {
    const int wave = threadIdx.x >> 5;
    const int lane = threadIdx.x & 31;
    const int lm   = lane & 15;   // M row within tile (A), N col within tile (B/C/D)
    const int lg   = lane >> 4;   // K-group select
    const int m0   = blockIdx.x * 128 + wave * 16;

    const float4* __restrict__ xrow = (const float4*)(x + (size_t)(m0 + lm) * DDIM);

    v8f acc[4] = {v8f{}, v8f{}, v8f{}, v8f{}};

    if constexpr (USE_TDM) {
        __shared__ __attribute__((aligned(32))) _Float16 bstage[2][ODIM][GROUP_ELEMS]; // 64 KB

        if (wave == 0) tdm_load_group(coeffh, 0, &bstage[0][0][0]);

        for (int g = 0; g < NGROUPS; ++g) {
            if (wave == 0) {
                if (g + 1 < NGROUPS) {
                    tdm_load_group(coeffh, g + 1, &bstage[(g + 1) & 1][0][0]);
                    // two in flight; oldest (group g) must be complete
                    __builtin_amdgcn_s_wait_tensorcnt(1);
                } else {
                    __builtin_amdgcn_s_wait_tensorcnt(0);
                }
            }
            __syncthreads();   // TDM data for group g visible to all waves

#pragma unroll
            for (int cc = 0; cc < GROUP_CHUNKS; ++cc) {
                const int c = g * GROUP_CHUNKS + cc;
                __builtin_prefetch(xrow + c + 4, 0, 1);   // global_prefetch_b8
                float4 xv = xrow[c];
                float da = lg ? xv.z : xv.x;              // d = 4c + 2*lg
                float db = lg ? xv.w : xv.y;              // d = 4c + 2*lg + 1
                float basA[8], basB[8];
                bspline8(da, basA);
                bspline8(db, basB);
                v16h a;
#pragma unroll
                for (int i = 0; i < 8; ++i) {
                    a[i]     = (_Float16)basA[i];
                    a[8 + i] = (_Float16)basB[i];
                }
#pragma unroll
                for (int nt = 0; nt < 4; ++nt) {
                    const v16h b = *(const v16h*)(&bstage[g & 1][nt * 16 + lm][cc * 32 + lg * 16]);
                    acc[nt] = __builtin_amdgcn_wmma_f32_16x16x32_f16(
                        false, a, false, b, (short)0, acc[nt], false, false);
                }
            }
            __syncthreads();   // group g consumed; buffer may be overwritten
        }
    } else {
        // Fallback: B fragments straight from fp32 coeff (converted in-flight).
        for (int c = 0; c < KDIM / 32; ++c) {
            const int kc = c * 32;
            __builtin_prefetch(xrow + c + 4, 0, 1);
            float4 xv = xrow[c];
            float da = lg ? xv.z : xv.x;
            float db = lg ? xv.w : xv.y;
            float basA[8], basB[8];
            bspline8(da, basA);
            bspline8(db, basB);
            v16h a;
#pragma unroll
            for (int i = 0; i < 8; ++i) {
                a[i]     = (_Float16)basA[i];
                a[8 + i] = (_Float16)basB[i];
            }
#pragma unroll
            for (int nt = 0; nt < 4; ++nt) {
                const size_t off = (size_t)(nt * 16 + lm) * KDIM + (size_t)kc + (size_t)lg * 16;
                const float4* p = (const float4*)(coeff + off);
                float4 f0 = p[0], f1 = p[1], f2 = p[2], f3 = p[3];
                v16h b;
                b[0]  = (_Float16)f0.x; b[1]  = (_Float16)f0.y;
                b[2]  = (_Float16)f0.z; b[3]  = (_Float16)f0.w;
                b[4]  = (_Float16)f1.x; b[5]  = (_Float16)f1.y;
                b[6]  = (_Float16)f1.z; b[7]  = (_Float16)f1.w;
                b[8]  = (_Float16)f2.x; b[9]  = (_Float16)f2.y;
                b[10] = (_Float16)f2.z; b[11] = (_Float16)f2.w;
                b[12] = (_Float16)f3.x; b[13] = (_Float16)f3.y;
                b[14] = (_Float16)f3.z; b[15] = (_Float16)f3.w;
                acc[nt] = __builtin_amdgcn_wmma_f32_16x16x32_f16(
                    false, a, false, b, (short)0, acc[nt], false, false);
            }
        }
    }

    // Epilogue: C/D layout -> VGPR v: lanes 0-15 hold M=v, lanes 16-31 hold M=v+8.
#pragma unroll
    for (int nt = 0; nt < 4; ++nt) {
        const int n  = nt * 16 + lm;
        const float bv = bias[n];
#pragma unroll
        for (int v = 0; v < 8; ++v) {
            const int r = m0 + lg * 8 + v;
            out[(size_t)r * ODIM + n] = acc[nt][v] + bv;
        }
    }
}

extern "C" void kernel_launch(void* const* d_in, const int* in_sizes, int n_in,
                              void* d_out, int out_size, void* d_ws, size_t ws_size,
                              hipStream_t stream) {
    const float* x     = (const float*)d_in[0];   // [16384, 512]
    const float* coeff = (const float*)d_in[1];   // [64, 512, 8]
    const float* bias  = (const float*)d_in[2];   // [64]
    // d_in[3] = knots: fixed linspace(-1,1,12), folded into constants.
    float* out = (float*)d_out;                   // [16384, 64]

    const size_t needH = (size_t)ODIM * (size_t)KDIM * 2u;  // f16 coeff staging
    if (ws_size >= needH) {
        _Float16* ch = (_Float16*)d_ws;
        kan_cvt_coeff_f16<<<(ODIM * KDIM / 4) / 256, 256, 0, stream>>>(
            (const float4*)coeff, (v4h*)ch);
        kan_wmma_kernel<true><<<BDIM / 128, 256, 0, stream>>>(x, coeff, ch, bias, out);
    } else {
        kan_wmma_kernel<false><<<BDIM / 128, 256, 0, stream>>>(x, coeff, nullptr, bias, out);
    }
}
